// GAT_39238821216303
// MI455X (gfx1250) — matrix operand order repaired and checked by reading, not compiled, not run on hardware
//
#include <hip/hip_runtime.h>
#include <hip/hip_bf16.h>

typedef __attribute__((ext_vector_type(16))) _Float16 v16h;
typedef __attribute__((ext_vector_type(8)))  float    v8f;

#define HD    128   // HEADS*OUT
#define EIN   32
#define NHEAD 2

// ---------- helpers ----------
__device__ __forceinline__ unsigned f2key(float f) {
  unsigned u = __float_as_uint(f);
  return (u & 0x80000000u) ? ~u : (u | 0x80000000u);   // order-preserving key
}
__device__ __forceinline__ float key2f(unsigned k) {
  unsigned u = (k & 0x80000000u) ? (k & 0x7FFFFFFFu) : ~k;
  return __uint_as_float(u);
}
__device__ __forceinline__ float lrelu(float v) { return v > 0.f ? v : 0.2f * v; }

// ---------- kernel 0: pack a row-major f32 weight [K x 128] into f16 WMMA-B
// fragments. Layout: Wp[((c*nq + q)*32 + lane)*16 + i] = W[q*32 + (i<8 ? 8*(lane>>4) + i : 16 + 8*(lane>>4) + i-8)][c*16 + (lane&15)]
// so each wave's B operand is one contiguous 32-byte load per lane. ----------
__global__ void k_pack_w(const float* __restrict__ W, _Float16* __restrict__ Wp, int nq) {
  const int t = blockIdx.x * blockDim.x + threadIdx.x;
  if (t >= 8 * nq * 512) return;
  const int i  = t & 15;
  const int l  = (t >> 4) & 31;
  const int cq = t >> 9;
  const int q  = cq % nq;
  const int c  = cq / nq;
  const int k  = q * 32 + ((i < 8) ? (8 * (l >> 4) + i) : (16 + 8 * (l >> 4) + (i - 8)));
  const int col = c * 16 + (l & 15);
  Wp[t] = (_Float16)W[(size_t)k * HD + col];
}

// ---------- kernel 1: h = x @ W_gat  (WMMA, A staged via LDS, B pre-packed) ----------
// block = 256 (8 waves); block -> one 16-row stripe; wave w -> 16-col tile w.
__global__ void __launch_bounds__(256)
k_node_gemm(const float* __restrict__ x, const _Float16* __restrict__ Wp,
            float* __restrict__ hbuf) {
  __shared__ __align__(32) _Float16 aF[4][32][16];   // 4 K-chunks of fragment-ordered A
  const int row0 = blockIdx.x * 16;
  const int tid  = threadIdx.x;
  if (tid < 128) {           // waves 0-3 fill LDS (wave-uniform branch)
    const int q = tid >> 5, l = tid & 31, m = l & 15, half = l >> 4;
#pragma unroll
    for (int i = 0; i < 8; ++i) {
      // A layout (ISA 7.12.2, 16-bit A 16x32): V0-3 -> K 0..7 (+8*half), V4-7 -> K 16..23 (+8*half)
      const int ka = ((i < 4) ? (2 * i) : (16 + 2 * (i - 4))) + 8 * half;
      const float2 av = *(const float2*)(x + (size_t)(row0 + m) * HD + q * 32 + ka);
      aF[q][l][2 * i]     = (_Float16)av.x;
      aF[q][l][2 * i + 1] = (_Float16)av.y;
    }
  }
  __syncthreads();
  const int wv = tid >> 5, lane = tid & 31;
  v8f acc = {};
#pragma unroll
  for (int q = 0; q < 4; ++q) {
    const v16h a = *(const v16h*)(&aF[q][lane][0]);
    const v16h b = *(const v16h*)(Wp + ((size_t)(wv * 4 + q) * 32 + lane) * 16);
    acc = __builtin_amdgcn_wmma_f32_16x16x32_f16(false, a, false, b, (short)0, acc,
                                                 false, false);
  }
  const int n = lane & 15, ro = (lane >> 4) * 8, col0 = wv * 16;
#pragma unroll
  for (int r = 0; r < 8; ++r)
    hbuf[(size_t)(row0 + r + ro) * HD + col0 + n] = acc[r];
}

// ---------- kernel 2: per-node attention scalars ----------
__global__ void k_asd(const float* __restrict__ hbuf,
                      const float* __restrict__ att_src,
                      const float* __restrict__ att_dst,
                      float* __restrict__ a_s, float* __restrict__ a_d, int N) {
  const int t = blockIdx.x * blockDim.x + threadIdx.x;
  if (t >= N * NHEAD) return;
  const int n = t >> 1, hh = t & 1;
  const float* hr = hbuf + (size_t)n * HD + hh * 64;
  const float* as = att_src + hh * 64;
  const float* ad = att_dst + hh * 64;
  float s1 = 0.f, s2 = 0.f;
#pragma unroll 8
  for (int d = 0; d < 64; ++d) { const float v = hr[d]; s1 += v * as[d]; s2 += v * ad[d]; }
  a_s[t] = s1; a_d[t] = s2;
}

// ---------- kernel 3: segment-sum edge_attr onto src ----------
__global__ void k_edge_sum(const int* __restrict__ ei, const float* __restrict__ ea,
                           float* __restrict__ esum, int E) {
  const int t = blockIdx.x * blockDim.x + threadIdx.x;
  if (t >= E * EIN) return;
  const int e = t >> 5, j = t & 31;
  atomicAdd(&esum[(size_t)ei[e] * EIN + j], ea[t]);
}

// ---------- kernel 4: attention pass A (segment max + out-degree) ----------
__global__ void k_attn_max(const int* __restrict__ ei, const float* __restrict__ a_s,
                           const float* __restrict__ a_d, unsigned* __restrict__ mkey,
                           unsigned* __restrict__ deg, int E, int N) {
  const int t = blockIdx.x * blockDim.x + threadIdx.x;
  if (t >= E + N) return;
  int s, d;
  if (t < E) { s = ei[t]; d = ei[E + t]; atomicAdd(&deg[s], 1u); }
  else       { s = t - E; d = s; }
#pragma unroll
  for (int h = 0; h < NHEAD; ++h) {
    const float v = lrelu(a_s[s * NHEAD + h] + a_d[d * NHEAD + h]);
    atomicMax(&mkey[d * NHEAD + h], f2key(v));
  }
}

// ---------- kernel 5: attention pass B (denominator) ----------
__global__ void k_attn_den(const int* __restrict__ ei, const float* __restrict__ a_s,
                           const float* __restrict__ a_d, const unsigned* __restrict__ mkey,
                           float* __restrict__ denom, int E, int N) {
  const int t = blockIdx.x * blockDim.x + threadIdx.x;
  if (t >= E + N) return;
  int s, d;
  if (t < E) { s = ei[t]; d = ei[E + t]; } else { s = t - E; d = s; }
#pragma unroll
  for (int h = 0; h < NHEAD; ++h) {
    const float v = lrelu(a_s[s * NHEAD + h] + a_d[d * NHEAD + h]);
    atomicAdd(&denom[d * NHEAD + h], __expf(v - key2f(mkey[d * NHEAD + h])));
  }
}

// ---------- kernel 6: attention pass C (alpha-weighted message scatter) ----------
// one 128-thread block per (edge|self-loop); feature f = threadIdx.x
__global__ void __launch_bounds__(128)
k_attn_msg(const int* __restrict__ ei, const float* __restrict__ a_s,
           const float* __restrict__ a_d, const unsigned* __restrict__ mkey,
           const float* __restrict__ denom, const float* __restrict__ hbuf,
           float* __restrict__ node_feat, int E) {
  const int t = blockIdx.x;
  const int f = threadIdx.x;
  int s, d;
  if (t < E) { s = ei[t]; d = ei[E + t]; } else { s = t - E; d = s; }
  __shared__ float alpha[NHEAD];
  if (f < NHEAD) {
    const float v = lrelu(a_s[s * NHEAD + f] + a_d[d * NHEAD + f]);
    alpha[f] = __expf(v - key2f(mkey[d * NHEAD + f])) / denom[d * NHEAD + f];
  }
  __syncthreads();
  atomicAdd(&node_feat[(size_t)d * HD + f], alpha[f >> 6] * hbuf[(size_t)s * HD + f]);
}

// ---------- kernel 7: node_feat += edge_sum @ W_edge  (WMMA, C-accumulate) ----------
__global__ void __launch_bounds__(256)
k_edge_gemm(const float* __restrict__ esum, const _Float16* __restrict__ Wp,
            float* __restrict__ nf) {
  __shared__ __align__(32) _Float16 aF[32][16];
  const int row0 = blockIdx.x * 16;
  const int tid  = threadIdx.x;
  if (tid < 32) {            // wave 0 fills LDS (wave-uniform branch)
    const int l = tid, m = l & 15, half = l >> 4;
#pragma unroll
    for (int i = 0; i < 8; ++i) {
      const int ka = ((i < 4) ? (2 * i) : (16 + 2 * (i - 4))) + 8 * half;
      const float2 av = *(const float2*)(esum + (size_t)(row0 + m) * EIN + ka);
      aF[l][2 * i]     = (_Float16)av.x;
      aF[l][2 * i + 1] = (_Float16)av.y;
    }
  }
  __syncthreads();
  const int wv = tid >> 5, lane = tid & 31, n = lane & 15, ro = (lane >> 4) * 8;
  const int col0 = wv * 16;
  v8f acc;
#pragma unroll
  for (int r = 0; r < 8; ++r)
    acc[r] = nf[(size_t)(row0 + r + ro) * HD + col0 + n];
  const v16h a = *(const v16h*)(&aF[lane][0]);
  const v16h b = *(const v16h*)(Wp + ((size_t)wv * 32 + lane) * 16);
  acc = __builtin_amdgcn_wmma_f32_16x16x32_f16(false, a, false, b, (short)0, acc,
                                               false, false);
#pragma unroll
  for (int r = 0; r < 8; ++r)
    nf[(size_t)(row0 + r + ro) * HD + col0 + n] = acc[r];
}

// ---------- kernel 8: add biases, global-add-pool per graph ----------
__global__ void k_pool(const float* __restrict__ node_feat, const unsigned* __restrict__ deg,
                       const float* __restrict__ b_gat, const float* __restrict__ b_edge,
                       const int* __restrict__ batch, float* __restrict__ pooled, int N) {
  const int t = blockIdx.x * blockDim.x + threadIdx.x;
  if (t >= N * HD) return;
  const int n = t >> 7, f = t & 127;
  const float c = node_feat[t] + b_gat[f] + (float)deg[n] * b_edge[f];
  atomicAdd(&pooled[(size_t)batch[n] * HD + f], c);
}

// ---------- kernel 9: MLP head (tiny, scalar) ----------
__global__ void __launch_bounds__(128)
k_mlp(const float* __restrict__ pooled,
      const float* __restrict__ W1, const float* __restrict__ b1,
      const float* __restrict__ W2, const float* __restrict__ b2,
      const float* __restrict__ W3, const float* __restrict__ b3,
      float* __restrict__ out) {
  const int g = blockIdx.x, t = threadIdx.x;
  __shared__ float p[HD], z1[100], z2[25];
  p[t] = pooled[(size_t)g * HD + t];
  __syncthreads();
  if (t < 100) {
    float acc = b1[t];
    for (int k = 0; k < HD; ++k) acc += p[k] * W1[k * 100 + t];
    z1[t] = fmaxf(acc, 0.f);
  }
  __syncthreads();
  if (t < 25) {
    float acc = b2[t];
    for (int k = 0; k < 100; ++k) acc += z1[k] * W2[k * 25 + t];
    z2[t] = fmaxf(acc, 0.f);
  }
  __syncthreads();
  if (t < 2) {
    float acc = b3[t];
    for (int k = 0; k < 25; ++k) acc += z2[k] * W3[k * 2 + t];
    out[g * 2 + t] = acc;
  }
}

extern "C" void kernel_launch(void* const* d_in, const int* in_sizes, int n_in,
                              void* d_out, int out_size, void* d_ws, size_t ws_size,
                              hipStream_t stream) {
  const float* x       = (const float*)d_in[0];
  const int*   ei      = (const int*)d_in[1];
  const float* ea      = (const float*)d_in[2];
  const int*   batch   = (const int*)d_in[3];
  const float* W_gat   = (const float*)d_in[4];
  const float* att_src = (const float*)d_in[5];
  const float* att_dst = (const float*)d_in[6];
  const float* b_gat   = (const float*)d_in[7];
  const float* W_edge  = (const float*)d_in[8];
  const float* b_edge  = (const float*)d_in[9];
  const float* W1 = (const float*)d_in[10];
  const float* b1 = (const float*)d_in[11];
  const float* W2 = (const float*)d_in[12];
  const float* b2 = (const float*)d_in[13];
  const float* W3 = (const float*)d_in[14];
  const float* b3 = (const float*)d_in[15];
  float* out = (float*)d_out;

  const int N = in_sizes[0] / HD;      // 50000
  const int E = in_sizes[1] / 2;       // 1600000
  const int G = out_size / 2;          // 256

  // -------- workspace layout (f32 elements) --------
  float* ws = (float*)d_ws;
  const size_t o_h  = 0;                               // h: N x 128 (no init)
  const size_t o_as = o_h  + (size_t)N * HD;           // a_s: N x 2
  const size_t o_ad = o_as + (size_t)N * NHEAD;        // a_d: N x 2
  const size_t o_mk = o_ad + (size_t)N * NHEAD;        // -- zero region from here --
  const size_t o_dn = o_mk + (size_t)N * NHEAD;        // denom
  const size_t o_dg = o_dn + (size_t)N * NHEAD;        // out-degree (u32)
  const size_t o_es = o_dg + (size_t)N;                // edge_sum: N x 32
  const size_t o_nf = o_es + (size_t)N * EIN;          // node_feat: N x 128
  const size_t o_pl = o_nf + (size_t)N * HD;           // pooled: G x 128
  const size_t o_end = o_pl + (size_t)G * HD;          // -- end of zero region --
  const size_t o_wg16 = o_end;                         // packed W_gat: 16384 halves
  const size_t o_we16 = o_wg16 + 8192;                 // packed W_edge: 4096 halves

  float*     hbuf  = ws + o_h;
  float*     a_s   = ws + o_as;
  float*     a_d   = ws + o_ad;
  unsigned*  mkey  = (unsigned*)(ws + o_mk);
  float*     denom = ws + o_dn;
  unsigned*  deg   = (unsigned*)(ws + o_dg);
  float*     esum  = ws + o_es;
  float*     nfeat = ws + o_nf;
  float*     pool  = ws + o_pl;
  _Float16*  wg16  = (_Float16*)(ws + o_wg16);
  _Float16*  we16  = (_Float16*)(ws + o_we16);

  // zero all accumulator buffers in one async memset (graph-capturable)
  hipMemsetAsync(ws + o_mk, 0, (o_end - o_mk) * sizeof(float), stream);

  // 0) pack weights into f16 WMMA-B fragment order
  k_pack_w<<<(8 * 4 * 512 + 255) / 256, 256, 0, stream>>>(W_gat, wg16, 4);
  k_pack_w<<<(8 * 1 * 512 + 255) / 256, 256, 0, stream>>>(W_edge, we16, 1);
  // 1) h = x @ W_gat            (WMMA)
  k_node_gemm<<<N / 16, 256, 0, stream>>>(x, wg16, hbuf);
  // 2) attention scalars a_s, a_d
  k_asd<<<(N * NHEAD + 255) / 256, 256, 0, stream>>>(hbuf, att_src, att_dst, a_s, a_d, N);
  // 3) segment-sum edge_attr onto src nodes
  k_edge_sum<<<(E * EIN + 255) / 256, 256, 0, stream>>>(ei, ea, esum, E);
  // 4) segment max over dst (incl. self-loops) + out-degree
  k_attn_max<<<(E + N + 255) / 256, 256, 0, stream>>>(ei, a_s, a_d, mkey, deg, E, N);
  // 5) softmax denominator
  k_attn_den<<<(E + N + 255) / 256, 256, 0, stream>>>(ei, a_s, a_d, mkey, denom, E, N);
  // 6) alpha-weighted message scatter into node_feat
  k_attn_msg<<<E + N, 128, 0, stream>>>(ei, a_s, a_d, mkey, denom, hbuf, nfeat, E);
  // 7) node_feat += edge_sum @ W_edge   (WMMA, C-accumulate)
  k_edge_gemm<<<N / 16, 256, 0, stream>>>(esum, we16, nfeat);
  // 8) biases + global add pool
  k_pool<<<(N * HD + 255) / 256, 256, 0, stream>>>(nfeat, deg, b_gat, b_edge, batch, pool, N);
  // 9) MLP head
  k_mlp<<<G, 128, 0, stream>>>(pool, W1, b1, W2, b2, W3, b3, out);
}
